// SimpleMoE_16655883174568
// MI455X (gfx1250) — matrix operand order repaired and checked by reading, not compiled, not run on hardware
//
#include <hip/hip_runtime.h>
#include <hip/hip_bf16.h>
#include <math.h>

// Problem constants (from reference): B=4, T=2048, D=1024, DFF=4096, E=8
#define N_TOK   8192
#define D_DIM   1024
#define DFF_DIM 4096
#define E_NUM   8
#define TILE_M  32               // tokens per block (2 WMMA M-tiles)
#define CHUNK   128              // DFF columns processed per iteration
#define MAXTILES (N_TOK / TILE_M + E_NUM)   // 264 >= sum_e ceil(count_e/32)
#define SMEM_BYTES (TILE_M * D_DIM * 2 + TILE_M * CHUNK * 2)  // 73728 B

typedef __attribute__((ext_vector_type(16))) _Float16 v16h;
typedef __attribute__((ext_vector_type(8)))  _Float16 v8h;
typedef __attribute__((ext_vector_type(4)))  _Float16 v4h;
typedef __attribute__((ext_vector_type(8)))  float    v8f;

__device__ __forceinline__ float gelu_exact(float v) {
    return 0.5f * v * (1.0f + erff(v * 0.70710678118654752440f));
}

// A fragment (16x32 f16, MxK): lane l holds row m=l&15; VGPR halves cover
// K = kb + (l>>4)*16 + {0..7} and K = kb + (l>>4)*16 + 8 + {0..7}.
// Both 8-half groups are 16B-contiguous -> two ds_load_b128 from an LDS row.
__device__ __forceinline__ v16h load_a_frag(const _Float16* __restrict__ row,
                                            int K0, int K1) {
    v8h t0 = *(const v8h*)(row + K0);
    v8h t1 = *(const v8h*)(row + K1);
    v16h a;
#pragma unroll
    for (int i = 0; i < 8; ++i) { a[i] = t0[i]; a[8 + i] = t1[i]; }
    return a;
}

// B fragment (32x16 f16, KxN) from fp32 weights: lane l holds column n;
// 16 consecutive K rows starting at koff=(l>>4)*16 relative base, stride ld.
__device__ __forceinline__ v16h load_b_frag_f32(const float* __restrict__ base,
                                                int ld) {
    v16h b;
#pragma unroll
    for (int i = 0; i < 16; ++i) b[i] = (_Float16)base[(size_t)i * (size_t)ld];
    return b;
}

// ---------------------------------------------------------------------------
// Kernel 1: gating.  One wave32 per token; 8 tokens per 256-thread block.
// ---------------------------------------------------------------------------
__global__ void __launch_bounds__(256)
moe_gate_kernel(const float* __restrict__ x, const float* __restrict__ gW,
                const float* __restrict__ gb, int* __restrict__ top1) {
    int tok  = blockIdx.x * 8 + (threadIdx.x >> 5);
    int lane = threadIdx.x & 31;
    if (tok >= N_TOK) return;
    float acc[E_NUM];
#pragma unroll
    for (int e = 0; e < E_NUM; ++e) acc[e] = 0.0f;
    const float* xr = x + (size_t)tok * D_DIM;
    for (int k = lane; k < D_DIM; k += 32) {
        float xv = xr[k];
        const float4* g = (const float4*)(gW + (size_t)k * E_NUM);
        float4 g0 = g[0], g1 = g[1];
        acc[0] += xv * g0.x; acc[1] += xv * g0.y;
        acc[2] += xv * g0.z; acc[3] += xv * g0.w;
        acc[4] += xv * g1.x; acc[5] += xv * g1.y;
        acc[6] += xv * g1.z; acc[7] += xv * g1.w;
    }
#pragma unroll
    for (int off = 16; off > 0; off >>= 1) {
#pragma unroll
        for (int e = 0; e < E_NUM; ++e) acc[e] += __shfl_xor(acc[e], off, 32);
    }
    if (lane == 0) {
        float best = acc[0] + gb[0];
        int be = 0;
#pragma unroll
        for (int e = 1; e < E_NUM; ++e) {
            float v = acc[e] + gb[e];
            if (v > best) { best = v; be = e; }   // first-max, matches argmax
        }
        top1[tok] = be;
    }
}

// ---------------------------------------------------------------------------
// Kernel 2: histogram + offsets + tile offsets (single block).
// ---------------------------------------------------------------------------
__global__ void __launch_bounds__(256)
moe_prep_kernel(const int* __restrict__ top1, int* __restrict__ counts,
                int* __restrict__ offsets, int* __restrict__ tileoff,
                int* __restrict__ cursor) {
    __shared__ int sc[E_NUM];
    if (threadIdx.x < E_NUM) sc[threadIdx.x] = 0;
    __syncthreads();
    for (int i = threadIdx.x; i < N_TOK; i += 256) atomicAdd(&sc[top1[i]], 1);
    __syncthreads();
    if (threadIdx.x == 0) {
        int off = 0, toff = 0;
        for (int e = 0; e < E_NUM; ++e) {
            int ce = sc[e];
            counts[e]  = ce;
            offsets[e] = off;
            tileoff[e] = toff;
            cursor[e]  = 0;
            off  += ce;
            toff += (ce + TILE_M - 1) / TILE_M;
        }
        offsets[E_NUM] = off;
        tileoff[E_NUM] = toff;
    }
}

// ---------------------------------------------------------------------------
// Kernel 3: scatter token ids into per-expert contiguous lists.
// ---------------------------------------------------------------------------
__global__ void __launch_bounds__(256)
moe_scatter_kernel(const int* __restrict__ top1, const int* __restrict__ offsets,
                   int* __restrict__ cursor, int* __restrict__ sorted) {
    int i = blockIdx.x * 256 + threadIdx.x;
    if (i >= N_TOK) return;
    int e = top1[i];
    int pos = atomicAdd(&cursor[e], 1);
    sorted[offsets[e] + pos] = i;
}

// ---------------------------------------------------------------------------
// Kernel 4: fused expert FFN.  One block = 32 gathered tokens of one expert.
// 8 waves.  X tile staged f16 in dynamic LDS (64 KB); H chunk (32x128) staged
// f16 (8 KB); out tile (32x1024) lives in v8f WMMA accumulators (16 per wave).
// Each B fragment (16 strided fp32 loads + cvt) is reused by 2 WMMAs.
// ---------------------------------------------------------------------------
__global__ void __launch_bounds__(256)
moe_ffn_kernel(const float* __restrict__ x,
               const float* __restrict__ W1, const float* __restrict__ b1,
               const float* __restrict__ W2, const float* __restrict__ b2,
               const int* __restrict__ sorted, const int* __restrict__ offsets,
               const int* __restrict__ tileoff, const int* __restrict__ counts,
               float* __restrict__ out) {
    extern __shared__ __align__(16) char smem[];
    _Float16* Xs = (_Float16*)smem;                 // [TILE_M][D_DIM]
    _Float16* Hs = Xs + TILE_M * D_DIM;             // [TILE_M][CHUNK]
    __shared__ int stok[TILE_M];
    __shared__ int sexp, sbase;

    const int bid = blockIdx.x;
    if (threadIdx.x == 0) {
        int e = -1, tloc = 0;
        if (bid < tileoff[E_NUM]) {
            for (int i = 0; i < E_NUM; ++i)
                if (bid >= tileoff[i] && bid < tileoff[i + 1]) {
                    e = i; tloc = bid - tileoff[i];
                }
        }
        sexp = e; sbase = tloc * TILE_M;
    }
    __syncthreads();
    const int e = sexp;
    if (e < 0) return;                       // uniform across block
    if (threadIdx.x < TILE_M) {
        int r = sbase + threadIdx.x;
        stok[threadIdx.x] = (r < counts[e]) ? sorted[offsets[e] + r] : -1;
    }
    __syncthreads();

    // Stage X tile as f16 (zero rows for tail-padding tokens), float4 loads.
    for (int idx = threadIdx.x; idx < TILE_M * (D_DIM / 4); idx += 256) {
        int m  = idx >> 8;                   // D_DIM/4 == 256
        int k4 = idx & 255;
        int tok = stok[m];
        float4 v = make_float4(0.f, 0.f, 0.f, 0.f);
        if (tok >= 0)
            v = ((const float4*)(x + (size_t)tok * D_DIM))[k4];
        v4h h;
        h[0] = (_Float16)v.x; h[1] = (_Float16)v.y;
        h[2] = (_Float16)v.z; h[3] = (_Float16)v.w;
        *(v4h*)(Xs + (size_t)m * D_DIM + k4 * 4) = h;
    }
    __syncthreads();

    const int wave  = threadIdx.x >> 5;
    const int lane  = threadIdx.x & 31;
    const int ln    = lane & 15;
    const int hi    = lane >> 4;
    const int kbase = hi * 8;

    const float* W1e = W1 + (size_t)e * D_DIM * DFF_DIM;
    const float* W2e = W2 + (size_t)e * DFF_DIM * D_DIM;
    const float* b1e = b1 + (size_t)e * DFF_DIM;
    const float* b2e = b2 + (size_t)e * D_DIM;

    v8f acc2[2][8];                          // [M-tile][N-tile] output accums
#pragma unroll
    for (int mt = 0; mt < 2; ++mt)
#pragma unroll
        for (int t = 0; t < 8; ++t)
            acc2[mt][t] = (v8f){0.f,0.f,0.f,0.f,0.f,0.f,0.f,0.f};

    for (int c = 0; c < DFF_DIM / CHUNK; ++c) {
        // ---- phase A: H chunk = gelu(X @ W1[:, chunk] + b1) ----
        {
            const int nH = c * CHUNK + wave * 16 + ln;   // global DFF column
            v8f acc[2];
#pragma unroll
            for (int mt = 0; mt < 2; ++mt)
                acc[mt] = (v8f){0.f,0.f,0.f,0.f,0.f,0.f,0.f,0.f};
#pragma unroll 2
            for (int kb = 0; kb < D_DIM; kb += 32) {
                v16h b = load_b_frag_f32(
                    W1e + (size_t)(kb + hi * 16) * DFF_DIM + nH, DFF_DIM);
#pragma unroll
                for (int mt = 0; mt < 2; ++mt) {
                    v16h a = load_a_frag(Xs + (size_t)(mt * 16 + ln) * D_DIM,
                                         kb + hi * 16, kb + hi * 16 + 8);
                    acc[mt] = __builtin_amdgcn_wmma_f32_16x16x32_f16(
                        false, a, false, b, (short)0, acc[mt], false, false);
                }
            }
            float bias = b1e[nH];
#pragma unroll
            for (int mt = 0; mt < 2; ++mt)
#pragma unroll
                for (int r = 0; r < 8; ++r) {
                    float v = gelu_exact(acc[mt][r] + bias);
                    Hs[(size_t)(mt * 16 + r + hi * 8) * CHUNK +
                       wave * 16 + ln] = (_Float16)v;
                }
        }
        __syncthreads();
        // ---- phase B: out += H_chunk @ W2[chunk rows, :] ----
#pragma unroll 1
        for (int kb = 0; kb < CHUNK; kb += 32) {
            v16h a0 = load_a_frag(Hs + (size_t)ln * CHUNK,
                                  kb + hi * 16, kb + hi * 16 + 8);
            v16h a1 = load_a_frag(Hs + (size_t)(16 + ln) * CHUNK,
                                  kb + hi * 16, kb + hi * 16 + 8);
            const int kg = c * CHUNK + kb + hi * 16;     // global DFF row
#pragma unroll
            for (int t = 0; t < 8; ++t) {
                int n = wave * 128 + t * 16 + ln;        // global D column
                v16h b = load_b_frag_f32(
                    W2e + (size_t)kg * D_DIM + n, D_DIM);
                acc2[0][t] = __builtin_amdgcn_wmma_f32_16x16x32_f16(
                    false, a0, false, b, (short)0, acc2[0][t], false, false);
                acc2[1][t] = __builtin_amdgcn_wmma_f32_16x16x32_f16(
                    false, a1, false, b, (short)0, acc2[1][t], false, false);
            }
        }
        __syncthreads();
    }

    // Epilogue: add b2, scatter rows back to their token slots.
#pragma unroll
    for (int t = 0; t < 8; ++t) {
        int n = wave * 128 + t * 16 + ln;
        float bias = b2e[n];
#pragma unroll
        for (int mt = 0; mt < 2; ++mt)
#pragma unroll
            for (int r = 0; r < 8; ++r) {
                int m = mt * 16 + r + hi * 8;
                int tok = stok[m];
                if (tok >= 0)
                    out[(size_t)tok * D_DIM + n] = acc2[mt][t][r] + bias;
            }
    }
}

// ---------------------------------------------------------------------------
extern "C" void kernel_launch(void* const* d_in, const int* in_sizes, int n_in,
                              void* d_out, int out_size, void* d_ws,
                              size_t ws_size, hipStream_t stream) {
    (void)in_sizes; (void)n_in; (void)out_size; (void)ws_size;
    const float* x      = (const float*)d_in[0];
    const float* gate_W = (const float*)d_in[1];
    const float* gate_b = (const float*)d_in[2];
    const float* W1     = (const float*)d_in[3];
    const float* b1     = (const float*)d_in[4];
    const float* W2     = (const float*)d_in[5];
    const float* b2     = (const float*)d_in[6];
    float* out = (float*)d_out;

    // Workspace layout (int units); ~66 KB total.
    int* ws      = (int*)d_ws;
    int* top1    = ws;              // N_TOK
    int* counts  = ws + 8192;       // 8
    int* offsets = ws + 8208;       // 9
    int* tileoff = ws + 8224;       // 9
    int* cursor  = ws + 8240;       // 8
    int* sorted  = ws + 8256;       // N_TOK

    moe_gate_kernel<<<N_TOK / 8, 256, 0, stream>>>(x, gate_W, gate_b, top1);
    moe_prep_kernel<<<1, 256, 0, stream>>>(top1, counts, offsets, tileoff, cursor);
    moe_scatter_kernel<<<N_TOK / 256, 256, 0, stream>>>(top1, offsets, cursor, sorted);
    moe_ffn_kernel<<<MAXTILES, 256, SMEM_BYTES, stream>>>(x, W1, b1, W2, b2,
                                                          sorted, offsets,
                                                          tileoff, counts, out);
}
